// CondTupleGPT_26628797235769
// MI455X (gfx1250) — compile-verified, matrix operand order checked by reading, not run on hardware
//
#include <hip/hip_runtime.h>
#include <hip/hip_bf16.h>
#include <math.h>

// ---------------------------------------------------------------------------
// CDNA5 / gfx1250 two-group GPT forward.
// - all matmuls on v_wmma_f32_16x16x32_bf16 (wave32)
// - weights pre-converted+transposed to bf16 [N,K]; activations kept bf16
// - GEMM staging uses global_load_async_to_lds_b128 (ASYNCcnt) with a
//   double-buffered LDS pipeline: one s_wait_asynccnt + one barrier per K-step
// ---------------------------------------------------------------------------

typedef __attribute__((ext_vector_type(16))) __bf16 v16bf;
typedef __attribute__((ext_vector_type(8)))  float  v8f;

static __device__ __forceinline__ __bf16 to_bf16(float x) { return (__bf16)x; }

static __device__ __forceinline__ v8f wmma_bf16(v16bf a, v16bf b, v8f c) {
  return __builtin_amdgcn_wmma_f32_16x16x32_bf16(false, a, false, b,
                                                 (short)0, c, false, false);
}

// Async DMA of 16 bytes global -> LDS, tracked by ASYNCcnt.
static __device__ __forceinline__ void async_copy_b128(void* lds_dst, const void* gsrc) {
  unsigned  ldsa = (unsigned)(size_t)lds_dst;            // low 32b = LDS offset
  unsigned long long ga = (unsigned long long)(size_t)gsrc;
  asm volatile("global_load_async_to_lds_b128 %0, %1, off"
               :: "v"(ldsa), "v"(ga) : "memory");
}
static __device__ __forceinline__ void wait_async() {
  asm volatile("s_wait_asynccnt 0x0" ::: "memory");
}

// 16x32 bf16 operand fragment from LDS tile [row][k0..31], `stride` elems/row.
// CDNA5 16-bit A/B striping: lane[3:0]=row, lane[4]=K-half; two 16B reads.
static __device__ __forceinline__ v16bf load_frag(const __bf16* base, int stride) {
  int l = threadIdx.x & 31;
  const __bf16* p = base + (l & 15) * stride + (l >> 4) * 8;
  v16bf f;
#pragma unroll
  for (int i = 0; i < 8; ++i) f[i] = p[i];
#pragma unroll
  for (int i = 0; i < 8; ++i) f[i + 8] = p[i + 16];
  return f;
}

// ---------------------------------------------------------------------------
// Weight convert + transpose: W fp32 [K,N] -> Wt bf16 [Npad,K], zero padded.
// ---------------------------------------------------------------------------
__global__ __launch_bounds__(256)
void ctg_wt_kernel(const float* __restrict__ W, __bf16* __restrict__ Wt,
                   int K, int N, int Npad) {
  __shared__ float tile[32][33];
  int n0 = blockIdx.x * 32, k0 = blockIdx.y * 32;
  int tx = threadIdx.x & 31, ty = threadIdx.x >> 5;
#pragma unroll
  for (int i = 0; i < 4; ++i) {
    int k = k0 + ty + i * 8, n = n0 + tx;
    tile[ty + i * 8][tx] = (n < N) ? W[(size_t)k * N + n] : 0.0f;
  }
  __syncthreads();
#pragma unroll
  for (int i = 0; i < 4; ++i) {
    int n = n0 + ty + i * 8, k = k0 + tx;
    Wt[(size_t)n * K + k] = to_bf16(tile[tx][ty + i * 8]);
  }
  (void)Npad;
}

// ---------------------------------------------------------------------------
// GEMM: C[M,N] = epi(A[M,K]bf16 @ Wt[N,K]bf16^T + bias + res)
// 256 threads = 8 waves; block tile 64x256; wave tile 32x64 (8 accumulators).
// Double-buffered LDS fed by async-to-LDS copies.
// ---------------------------------------------------------------------------
#define G_TM 64
#define G_TN 256
#define G_TK 32
#define G_LD 40   // LDS row stride in elems (32 + 8 pad; 80B, 16B-aligned)

__global__ __launch_bounds__(256)
void ctg_gemm_kernel(const __bf16* __restrict__ A, const __bf16* __restrict__ Wt,
                     const float* __restrict__ bias, const float* __restrict__ res,
                     void* __restrict__ Cv, int obf,
                     int M, int N, int K, int epi) {
  __shared__ __align__(16) __bf16 sA[2][G_TM * G_LD];
  __shared__ __align__(16) __bf16 sB[2][G_TN * G_LD];
  const int t  = threadIdx.x;
  const int m0 = blockIdx.y * G_TM;
  const int n0 = blockIdx.x * G_TN;
  const int w  = t >> 5;
  const int wm = (w >> 2) * 32;   // 0,32
  const int wn = (w & 3)  * 64;   // 0,64,128,192

  const int r  = t >> 2;          // 0..63
  const int cb = (t & 3) * 16;    // byte offset within a 64B tile row

  auto issue_tile = [&](int k0, int buf) {
    // A: 64 rows x 64B, one b128 per thread
    async_copy_b128((char*)sA[buf] + r * (G_LD * 2) + cb,
                    (const char*)(A + (size_t)(m0 + r) * K + k0) + cb);
    // B: 256 rows x 64B, four b128 per thread
#pragma unroll
    for (int i = 0; i < 4; ++i) {
      int rb = r + i * 64;
      async_copy_b128((char*)sB[buf] + rb * (G_LD * 2) + cb,
                      (const char*)(Wt + (size_t)(n0 + rb) * K + k0) + cb);
    }
  };

  v8f c0 = {}, c1 = {}, c2 = {}, c3 = {}, c4 = {}, c5 = {}, c6 = {}, c7 = {};
  const int nk = K / G_TK;

  issue_tile(0, 0);
  for (int i = 0; i < nk; ++i) {
    wait_async();        // my copies for tile i have landed in LDS
    __syncthreads();     // everyone's have
    if (i + 1 < nk) issue_tile((i + 1) * G_TK, (i + 1) & 1);

    const __bf16* pa = sA[i & 1];
    const __bf16* pb = sB[i & 1];
    v16bf a0 = load_frag(pa + wm * G_LD, G_LD);
    v16bf a1 = load_frag(pa + (wm + 16) * G_LD, G_LD);
    v16bf b0 = load_frag(pb + wn * G_LD, G_LD);
    v16bf b1 = load_frag(pb + (wn + 16) * G_LD, G_LD);
    v16bf b2 = load_frag(pb + (wn + 32) * G_LD, G_LD);
    v16bf b3 = load_frag(pb + (wn + 48) * G_LD, G_LD);

    c0 = wmma_bf16(a0, b0, c0);
    c1 = wmma_bf16(a0, b1, c1);
    c2 = wmma_bf16(a0, b2, c2);
    c3 = wmma_bf16(a0, b3, c3);
    c4 = wmma_bf16(a1, b0, c4);
    c5 = wmma_bf16(a1, b1, c5);
    c6 = wmma_bf16(a1, b2, c6);
    c7 = wmma_bf16(a1, b3, c7);
  }

  const int l  = t & 31;
  const int rr = (l >> 4) * 8;
  const int cc = l & 15;

  auto store_tile = [&](v8f acc, int mb, int nb) {
#pragma unroll
    for (int rq = 0; rq < 8; ++rq) {
      int row = m0 + wm + mb + rq + rr;
      int col = n0 + wn + nb + cc;
      if (col < N) {
        float v = acc[rq];
        if (bias) v += bias[col];
        if (res)  v += res[(size_t)row * N + col];
        if (epi == 1) v = 0.5f * v * (1.0f + erff(v * 0.70710678118f));
        if (obf) ((__bf16*)Cv)[(size_t)row * N + col] = to_bf16(v);
        else     ((float*) Cv)[(size_t)row * N + col] = v;
      }
    }
  };
  store_tile(c0, 0, 0);  store_tile(c1, 0, 16);
  store_tile(c2, 0, 32); store_tile(c3, 0, 48);
  store_tile(c4, 16, 0); store_tile(c5, 16, 16);
  store_tile(c6, 16, 32); store_tile(c7, 16, 48);
  (void)M;
}

// ---------------------------------------------------------------------------
// Flash attention (causal), bf16 I/O. 128 threads = 4 waves; each wave owns a
// 16-row query tile; block shares staged K/V per 32-key block.
// ---------------------------------------------------------------------------
__global__ __launch_bounds__(128)
void ctg_attn_kernel(const __bf16* __restrict__ Q, const __bf16* __restrict__ Kp,
                     const __bf16* __restrict__ Vp, __bf16* __restrict__ O,
                     int T_, int D_, int HD_) {
  __shared__ __align__(16) __bf16 sK[32 * 72];     // [key][hd]  stride 72
  __shared__ __align__(16) __bf16 sV[64 * 40];     // [hd][key]  stride 40
  __shared__ __align__(16) __bf16 sP[4][16 * 40];  // per-wave P tile

  const int t = threadIdx.x;
  const int w = t >> 5;
  const int l = t & 31;
  const int b = blockIdx.z, h = blockIdx.y;
  const int q0 = blockIdx.x * 64 + w * 16;

  const __bf16* Qb = Q  + (size_t)(b * T_) * D_ + h * HD_;
  const __bf16* Kb = Kp + (size_t)(b * T_) * D_ + h * HD_;
  const __bf16* Vb = Vp + (size_t)(b * T_) * D_ + h * HD_;

  v16bf qa0, qa1;
  {
    const __bf16* qp = Qb + (size_t)(q0 + (l & 15)) * D_;
    int khi = (l >> 4) * 8;
#pragma unroll
    for (int i = 0; i < 8; ++i) {
      qa0[i]     = qp[khi + i];
      qa0[i + 8] = qp[khi + 16 + i];
      qa1[i]     = qp[32 + khi + i];
      qa1[i + 8] = qp[32 + khi + 16 + i];
    }
  }

  v8f o0 = {}, o1 = {}, o2 = {}, o3 = {};
  float mrow[8], lrow[8];
#pragma unroll
  for (int rq = 0; rq < 8; ++rq) { mrow[rq] = -3.0e38f; lrow[rq] = 0.0f; }

  const int rr = (l >> 4) * 8;
  const int cc = l & 15;
  const int kend = blockIdx.x * 64 + 64;

  for (int kb = 0; kb < kend; kb += 32) {
    // K: 32x64 bf16 copied as 16B chunks (2 per thread)
#pragma unroll
    for (int i = 0; i < 2; ++i) {
      int e = t + i * 128;           // 0..255 chunks
      int row = e >> 3, ch = e & 7;  // 8 chunks per row
      *(uint4*)((char*)sK + row * 144 + ch * 16) =
          *(const uint4*)((const char*)(Kb + (size_t)(kb + row) * D_) + ch * 16);
    }
    // V: transpose [key][hd] -> [hd][key]
#pragma unroll
    for (int i = 0; i < 16; ++i) {
      int e = t + i * 128;
      int key = e >> 6, hd = e & 63;
      sV[hd * 40 + key] = Vb[(size_t)(kb + key) * D_ + hd];
    }
    __syncthreads();

    if (kb <= q0 + 15) {
      v16bf k0a = load_frag(sK, 72);
      v16bf k0b = load_frag(sK + 32, 72);
      v16bf k1a = load_frag(sK + 16 * 72, 72);
      v16bf k1b = load_frag(sK + 16 * 72 + 32, 72);

      v8f s0 = {}, s1 = {};
      s0 = wmma_bf16(qa0, k0a, s0);
      s0 = wmma_bf16(qa1, k0b, s0);
      s1 = wmma_bf16(qa0, k1a, s1);
      s1 = wmma_bf16(qa1, k1b, s1);

      __bf16* Pw = sP[w];
#pragma unroll
      for (int rq = 0; rq < 8; ++rq) {
        int qi = q0 + rq + rr;
        float v0 = s0[rq] * 0.125f;   // 1/sqrt(HD)
        float v1 = s1[rq] * 0.125f;
        if (kb + cc > qi)      v0 = -3.0e38f;
        if (kb + 16 + cc > qi) v1 = -3.0e38f;
        float mt = fmaxf(v0, v1);
#pragma unroll
        for (int off = 8; off >= 1; off >>= 1)
          mt = fmaxf(mt, __shfl_xor(mt, off, 32));
        float mnew  = fmaxf(mrow[rq], mt);
        float scale = __expf(mrow[rq] - mnew);
        float p0 = __expf(v0 - mnew);
        float p1 = __expf(v1 - mnew);
        float ps = p0 + p1;
#pragma unroll
        for (int off = 8; off >= 1; off >>= 1)
          ps += __shfl_xor(ps, off, 32);
        lrow[rq] = lrow[rq] * scale + ps;
        mrow[rq] = mnew;
        o0[rq] *= scale; o1[rq] *= scale; o2[rq] *= scale; o3[rq] *= scale;
        Pw[(rq + rr) * 40 + cc]      = to_bf16(p0);
        Pw[(rq + rr) * 40 + 16 + cc] = to_bf16(p1);
      }

      v16bf pa  = load_frag(Pw, 40);
      v16bf vf0 = load_frag(sV, 40);
      v16bf vf1 = load_frag(sV + 16 * 40, 40);
      v16bf vf2 = load_frag(sV + 32 * 40, 40);
      v16bf vf3 = load_frag(sV + 48 * 40, 40);
      o0 = wmma_bf16(pa, vf0, o0);
      o1 = wmma_bf16(pa, vf1, o1);
      o2 = wmma_bf16(pa, vf2, o2);
      o3 = wmma_bf16(pa, vf3, o3);
    }
    __syncthreads();
  }

  __bf16* Ob = O + (size_t)(b * T_) * D_ + h * HD_;
#pragma unroll
  for (int rq = 0; rq < 8; ++rq) {
    int qi = q0 + rq + rr;
    float inv = 1.0f / lrow[rq];
    Ob[(size_t)qi * D_ + cc]      = to_bf16(o0[rq] * inv);
    Ob[(size_t)qi * D_ + 16 + cc] = to_bf16(o1[rq] * inv);
    Ob[(size_t)qi * D_ + 32 + cc] = to_bf16(o2[rq] * inv);
    Ob[(size_t)qi * D_ + 48 + cc] = to_bf16(o3[rq] * inv);
  }
}

// ---------------------------------------------------------------------------
// LayerNorm (fp32 in, bf16 out): one wave32 per row.
// ---------------------------------------------------------------------------
__global__ __launch_bounds__(256)
void ctg_layernorm_kernel(const float* __restrict__ X, const float* __restrict__ g,
                          const float* __restrict__ bta, __bf16* __restrict__ Y, int Dd) {
  int w = threadIdx.x >> 5, l = threadIdx.x & 31;
  int row = blockIdx.x * 8 + w;
  const float* x = X + (size_t)row * Dd;
  float sum = 0.0f, sq = 0.0f;
  for (int i = l; i < Dd; i += 32) { float v = x[i]; sum += v; sq += v * v; }
#pragma unroll
  for (int off = 16; off >= 1; off >>= 1) {
    sum += __shfl_xor(sum, off, 32);
    sq  += __shfl_xor(sq,  off, 32);
  }
  float mean = sum / Dd;
  float var  = sq / Dd - mean * mean;
  float rs   = rsqrtf(var + 1e-5f);
  __bf16* y = Y + (size_t)row * Dd;
  for (int i = l; i < Dd; i += 32)
    y[i] = to_bf16((x[i] - mean) * rs * g[i] + bta[i]);
}

// ---------------------------------------------------------------------------
// Embedding sum + positional split; target-token add (fp32 residual stream).
// ---------------------------------------------------------------------------
__global__ void ctg_embed_kernel(const int* __restrict__ idx, const int* __restrict__ eidx,
                                 const int* __restrict__ Lc_p,
                                 const float* __restrict__ tok0, const float* __restrict__ tok1,
                                 const float* __restrict__ extra,
                                 const float* __restrict__ pos, const float* __restrict__ cpos,
                                 float* __restrict__ X, int n, int Dd, int T_) {
  int e = blockIdx.x * blockDim.x + threadIdx.x;
  if (e >= n) return;
  int row = e / Dd, d = e % Dd;
  int tpos = row % T_;
  int Lc = Lc_p[0];
  float p = (tpos < Lc) ? cpos[tpos * Dd + d] : pos[(tpos - Lc) * Dd + d];
  X[e] = tok0[(size_t)idx[row * 2] * Dd + d]
       + tok1[(size_t)idx[row * 2 + 1] * Dd + d]
       + extra[(size_t)eidx[row] * Dd + d] + p;
}

__global__ void ctg_add_tok_kernel(const int* __restrict__ tgt, const float* __restrict__ tok,
                                   float* __restrict__ X, int n, int Dd, int which) {
  int e = blockIdx.x * blockDim.x + threadIdx.x;
  if (e >= n) return;
  int row = e / Dd, d = e % Dd;
  X[e] += tok[(size_t)tgt[row * 2 + which] * Dd + d];
}

// ---------------------------------------------------------------------------
// Host orchestration
// ---------------------------------------------------------------------------
extern "C" void kernel_launch(void* const* d_in, const int* in_sizes, int n_in,
                              void* d_out, int out_size, void* d_ws, size_t ws_size,
                              hipStream_t stream) {
  (void)in_sizes; (void)n_in; (void)out_size; (void)ws_size;
  constexpr int Bc = 2, Tc = 1024, Dc = 512, Hc = 8, HDc = 64, Fc = 2048;
  constexpr int Mc = Bc * Tc;
  constexpr size_t MD = (size_t)Mc * Dc;
  const int V0 = 8192, V1 = 4097;
  const int NL[2] = {6, 2};

  const int* idx  = (const int*)d_in[0];
  const int* eidx = (const int*)d_in[1];
  const int* Lc   = (const int*)d_in[2];
  const int* tgt  = (const int*)d_in[3];
  const float* tok0  = (const float*)d_in[4];
  const float* tok1  = (const float*)d_in[5];
  const float* extra = (const float*)d_in[6];
  const float* pose  = (const float*)d_in[7];
  const float* cpose = (const float*)d_in[8];
  const float* toks[2] = {tok0, tok1};
  const int gbase[2] = {9, 28};

  // workspace layout
  char* p = (char*)d_ws;
  float*  xa  = (float*)p;  p += MD * 4;
  float*  xb  = (float*)p;  p += MD * 4;
  __bf16* hbf = (__bf16*)p; p += MD * 2;
  __bf16* qb  = (__bf16*)p; p += MD * 2;
  __bf16* kb  = (__bf16*)p; p += MD * 2;
  __bf16* vb  = (__bf16*)p; p += MD * 2;
  __bf16* yb  = (__bf16*)p; p += MD * 2;
  __bf16* m1b = (__bf16*)p; p += (size_t)Mc * Fc * 2;
  __bf16* wtb = (__bf16*)p; p += (size_t)8192 * 512 * 2;  // max Npad*K bf16

  auto gemm = [&](const __bf16* A, const float* W, const float* bias,
                  const float* res, void* C, int obf, int M_, int N_, int K_, int epi) {
    int Npad = (N_ + 255) & ~255;
    ctg_wt_kernel<<<dim3(Npad / 32, K_ / 32), dim3(256), 0, stream>>>(W, wtb, K_, N_, Npad);
    ctg_gemm_kernel<<<dim3(Npad / G_TN, M_ / G_TM), dim3(256), 0, stream>>>(
        A, wtb, bias, res, C, obf, M_, N_, K_, epi);
  };
  auto lnorm = [&](const float* X, const float* g, const float* b, __bf16* Y) {
    ctg_layernorm_kernel<<<dim3(Mc / 8), dim3(256), 0, stream>>>(X, g, b, Y, Dc);
  };

  {
    int n = (int)MD;
    ctg_embed_kernel<<<dim3((n + 255) / 256), dim3(256), 0, stream>>>(
        idx, eidx, Lc, tok0, tok1, extra, pose, cpose, xa, n, Dc, Tc);
  }

  float* out0 = (float*)d_out;
  float* out1 = out0 + (size_t)Mc * V0;
  float* outs[2] = {out0, out1};
  const int Vs[2] = {V0, V1};

  for (int g = 0; g < 2; ++g) {
    const int base = gbase[g];
    const float* ln1_g = (const float*)d_in[base + 0];
    const float* ln1_b = (const float*)d_in[base + 1];
    const float* Wq = (const float*)d_in[base + 2];
    const float* bq = (const float*)d_in[base + 3];
    const float* Wk = (const float*)d_in[base + 4];
    const float* bk = (const float*)d_in[base + 5];
    const float* Wv = (const float*)d_in[base + 6];
    const float* bv = (const float*)d_in[base + 7];
    const float* Wp = (const float*)d_in[base + 8];
    const float* bp = (const float*)d_in[base + 9];
    const float* ln2_g = (const float*)d_in[base + 10];
    const float* ln2_b = (const float*)d_in[base + 11];
    const float* W1 = (const float*)d_in[base + 12];
    const float* b1 = (const float*)d_in[base + 13];
    const float* W2 = (const float*)d_in[base + 14];
    const float* b2 = (const float*)d_in[base + 15];
    const float* hln_g = (const float*)d_in[base + 16];
    const float* hln_b = (const float*)d_in[base + 17];
    const float* Wh    = (const float*)d_in[base + 18];

    for (int layer = 0; layer < NL[g]; ++layer) {
      size_t oD  = (size_t)layer * Dc;
      size_t oDD = (size_t)layer * Dc * Dc;
      size_t oDF = (size_t)layer * Dc * Fc;
      size_t oF  = (size_t)layer * Fc;

      lnorm(xa, ln1_g + oD, ln1_b + oD, hbf);
      gemm(hbf, Wq + oDD, bq + oD, nullptr, qb, 1, Mc, Dc, Dc, 0);
      gemm(hbf, Wk + oDD, bk + oD, nullptr, kb, 1, Mc, Dc, Dc, 0);
      gemm(hbf, Wv + oDD, bv + oD, nullptr, vb, 1, Mc, Dc, Dc, 0);

      ctg_attn_kernel<<<dim3(Tc / 64, Hc, Bc), dim3(128), 0, stream>>>(
          qb, kb, vb, yb, Tc, Dc, HDc);

      gemm(yb, Wp + oDD, bp + oD, xa, xb, 0, Mc, Dc, Dc, 0);      // xb = xa + y@Wp+bp
      lnorm(xb, ln2_g + oD, ln2_b + oD, hbf);
      gemm(hbf, W1 + oDF, b1 + oF, nullptr, m1b, 1, Mc, Fc, Dc, 1);  // gelu -> bf16
      gemm(m1b, W2 + oDF, b2 + oD, xb, xa, 0, Mc, Dc, Fc, 0);     // xa = xb + mlp
    }

    lnorm(xa, hln_g, hln_b, hbf);
    gemm(hbf, Wh, nullptr, nullptr, outs[g], 0, Mc, Vs[g], Dc, 0);

    {
      int n = (int)MD;
      ctg_add_tok_kernel<<<dim3((n + 255) / 256), dim3(256), 0, stream>>>(
          tgt, toks[g], xa, n, Dc, g);
    }
  }
}
